// LSTM_84121229460130
// MI455X (gfx1250) — matrix operand order repaired and checked by reading, not compiled, not run on hardware
//
#include <hip/hip_runtime.h>
#include <hip/hip_bf16.h>

// LSTM forward for MI455X (gfx1250): bf16 WMMA recurrence, fused x-projection.
// T=512, B=128, E=256, Hd=512. Gates concatenated along N: [i|f|o|c] -> N=2048.
// K dimension concatenated: [x (256) | h (512)] -> K=768.

#define T_    512
#define B_    128
#define E_    256
#define HD_   512
#define KTOT  768          // E_ + HD_
#define NTOT  2048         // 4 * HD_
#define KC_N  24           // KTOT / 32 (K-chunks per WMMA)
#define ASTR  776          // padded LDS row stride (bf16) to break bank conflicts
#define ACCS  33           // padded accumulator row stride (f32)
#define ACCG  (32 * ACCS)  // per-gate accumulator slab (f32)

typedef __attribute__((ext_vector_type(16))) __bf16 v16bf;
typedef __attribute__((ext_vector_type(8)))  __bf16 v8bf;
typedef __attribute__((ext_vector_type(8)))  float  v8f;
typedef __attribute__((ext_vector_type(4)))  int    v4i;

typedef __attribute__((address_space(1))) v4i gv4i;  // global int4
typedef __attribute__((address_space(3))) v4i lv4i;  // LDS int4

#if __has_builtin(__builtin_amdgcn_global_load_async_to_lds_b128)
#define USE_ASYNC_LDS 1
#else
#define USE_ASYNC_LDS 0
#endif

static __device__ __forceinline__ unsigned short f32_to_bf16(float f) {
  union { float f; unsigned int u; } x; x.f = f;
  unsigned int u = x.u;
  unsigned int r = (u + 0x7fffu + ((u >> 16) & 1u)) >> 16;
  return (unsigned short)r;
}

// Stage one 16B chunk global -> LDS (async-to-LDS on CDNA5 if available).
static __device__ __forceinline__ void stage16(const unsigned short* g, unsigned short* l) {
#if USE_ASYNC_LDS
  __builtin_amdgcn_global_load_async_to_lds_b128(
      (gv4i*)(void*)g, (lv4i*)(void*)l, 0, 0);
#else
  *reinterpret_cast<v8bf*>(l) = *reinterpret_cast<const v8bf*>(g);
#endif
}

static __device__ __forceinline__ void stage_wait() {
#if USE_ASYNC_LDS
#if __has_builtin(__builtin_amdgcn_s_wait_asynccnt)
  __builtin_amdgcn_s_wait_asynccnt(0);
#else
  asm volatile("s_wait_asynccnt 0x0" ::: "memory");
#endif
#endif
}

// ---------------------------------------------------------------------------
// One-time: pack [Wx;Wh] for all 4 gates into WMMA B-fragment order (bf16).
// Record = 32 bytes per lane: ntile (16 cols) x kc (32 K) tile,
// lane = half*16 + n, contents K = kc*32 + half*16 + j (j=0..15, contiguous).
// ---------------------------------------------------------------------------
__global__ void pack_weights(const float* __restrict__ Wxi, const float* __restrict__ Whi,
                             const float* __restrict__ Wxf, const float* __restrict__ Whf,
                             const float* __restrict__ Wxo, const float* __restrict__ Who,
                             const float* __restrict__ Wxc, const float* __restrict__ Whc,
                             unsigned short* __restrict__ Wp) {
  int p = blockIdx.x * blockDim.x + threadIdx.x;
  if (p >= KTOT * NTOT) return;
  int j     = p & 15;
  int rec   = p >> 4;
  int lane  = rec & 31;
  int tmp   = rec >> 5;
  int kc    = tmp % KC_N;
  int ntile = tmp / KC_N;
  int k  = kc * 32 + ((lane >> 4) << 4) + j;
  int ng = ntile * 16 + (lane & 15);
  int gate = ng >> 9;
  int h    = ng & (HD_ - 1);
  const float* Wx = (gate == 0) ? Wxi : (gate == 1) ? Wxf : (gate == 2) ? Wxo : Wxc;
  const float* Wh = (gate == 0) ? Whi : (gate == 1) ? Whf : (gate == 2) ? Who : Whc;
  float v = (k < E_) ? Wx[k * HD_ + h] : Wh[(k - E_) * HD_ + h];
  Wp[p] = f32_to_bf16(v);
}

__global__ void pack_bias(const float* __restrict__ bi, const float* __restrict__ bf,
                          const float* __restrict__ bo, const float* __restrict__ bc,
                          float* __restrict__ bias) {
  int n = blockIdx.x * blockDim.x + threadIdx.x;
  if (n >= NTOT) return;
  int g = n >> 9, h = n & (HD_ - 1);
  const float* bp = (g == 0) ? bi : (g == 1) ? bf : (g == 2) ? bo : bc;
  bias[n] = bp[h];
}

__global__ void cvt_inputs(const float* __restrict__ X, unsigned short* __restrict__ Xbf, int N) {
  int i = blockIdx.x * blockDim.x + threadIdx.x;
  if (i < N) Xbf[i] = f32_to_bf16(X[i]);
}

__global__ void init_state(unsigned short* __restrict__ H0, unsigned short* __restrict__ H1,
                           float* __restrict__ C) {
  int i = blockIdx.x * blockDim.x + threadIdx.x;
  if (i < B_ * HD_) { H0[i] = 0; H1[i] = 0; C[i] = 0.f; }
}

// ---------------------------------------------------------------------------
// One LSTM timestep. Grid: (4 mTiles of 32 rows, 16 hTiles of 32 h), 256 thr.
// Wave w: gate = w>>1, sub = w&1 -> two 16x16 WMMA tiles (rows 0-15 / 16-31)
// at n = gate*512 + hTile*32 + sub*16, sharing each B fragment (halves W L2
// traffic vs one tile/wave). K = 768 in 24 chunks of 32.
// ---------------------------------------------------------------------------
__global__ __launch_bounds__(256) void lstm_step(
    const unsigned short* __restrict__ Xbf,
    const unsigned short* __restrict__ Wp,
    const float* __restrict__ bias,
    const unsigned short* __restrict__ Hread,
    unsigned short* __restrict__ Hwrite,
    float* __restrict__ Cst,
    float* __restrict__ out,
    float* __restrict__ outH,
    float* __restrict__ outC,
    int t) {
  // A slab (32 x ASTR bf16 = 49.6 KB) overlaid with accumulator slab
  // (4 x 32 x ACCS f32 = 16.9 KB); phases separated by barriers.
  __shared__ __align__(16) unsigned char smem[32 * ASTR * 2];
  unsigned short* Alds  = reinterpret_cast<unsigned short*>(smem);
  float*          accLd = reinterpret_cast<float*>(smem);

  const int tid   = threadIdx.x;
  const int lane  = tid & 31;
  const int wave  = tid >> 5;
  const int mTile = blockIdx.x;  // 0..3  (32 batch rows each)
  const int hTile = blockIdx.y;  // 0..15 (32 h each)

  // ---- Stage A tile (32 rows x 768 K, bf16) into LDS (16B chunks)
  {
    const unsigned short* Xsrc = Xbf + ((size_t)t * B_ + (size_t)mTile * 32) * E_;
#pragma unroll
    for (int i = 0; i < 4; ++i) {
      int e = (tid + i * 256) * 8;          // 1024 chunks (X part, contiguous)
      int m = e >> 8;                        // / E_
      int k = e & (E_ - 1);
      stage16(Xsrc + e, &Alds[m * ASTR + k]);
    }
    const unsigned short* Hsrc = Hread + (size_t)mTile * 32 * HD_;
#pragma unroll
    for (int i = 0; i < 8; ++i) {
      int e = (tid + i * 256) * 8;          // 2048 chunks (H part, contiguous)
      int m = e >> 9;                        // / HD_
      int k = e & (HD_ - 1);
      stage16(Hsrc + e, &Alds[m * ASTR + E_ + k]);
    }
  }
  stage_wait();
  __syncthreads();

  // ---- WMMA: two 16x16 f32 tiles per wave, shared B fragment
  const int gate  = wave >> 1;
  const int sub   = wave & 1;
  const int ntile = gate * 32 + hTile * 2 + sub;  // 0..127
  const int mrow  = lane & 15;
  const int half  = lane >> 4;

  v8f acc0 = {0.f, 0.f, 0.f, 0.f, 0.f, 0.f, 0.f, 0.f};
  v8f acc1 = {0.f, 0.f, 0.f, 0.f, 0.f, 0.f, 0.f, 0.f};
  const v16bf* Wrec =
      reinterpret_cast<const v16bf*>(Wp) + (size_t)ntile * (KC_N * 32) + lane;
  union { v16bf v; v8bf h[2]; } a0, a1;

  for (int kc = 0; kc < KC_N; ++kc) {
    const int kb = kc * 32;
    const v16bf bfrag = Wrec[kc * 32];  // 32B contiguous per lane, coalesced
    // A fragments: lane holds row r, K = kb + half*16 + [0..7] (VGPR 0-3)
    //                           and K = kb + half*16 + 8 + [0..7] (VGPR 4-7)
    a0.h[0] = *reinterpret_cast<const v8bf*>(&Alds[mrow * ASTR + kb + half * 16]);
    a0.h[1] = *reinterpret_cast<const v8bf*>(&Alds[mrow * ASTR + kb + 8 + half * 16]);
    a1.h[0] = *reinterpret_cast<const v8bf*>(&Alds[(mrow + 16) * ASTR + kb + half * 16]);
    a1.h[1] = *reinterpret_cast<const v8bf*>(&Alds[(mrow + 16) * ASTR + kb + 8 + half * 16]);
    acc0 = __builtin_amdgcn_wmma_f32_16x16x32_bf16(
        false, a0.v, false, bfrag, (short)0, acc0, false, false);
    acc1 = __builtin_amdgcn_wmma_f32_16x16x32_bf16(
        false, a1.v, false, bfrag, (short)0, acc1, false, false);
  }
  __syncthreads();  // everyone done reading Alds before overlaying accLd

  // ---- Scatter accumulators to LDS (C/D layout: VGPR r -> M = r + 8*half)
  {
    float* dst = &accLd[gate * ACCG];
    const int n = sub * 16 + (lane & 15);
#pragma unroll
    for (int r = 0; r < 8; ++r) {
      dst[(r + 8 * half) * ACCS + n]      = acc0[r];
      dst[(16 + r + 8 * half) * ACCS + n] = acc1[r];
    }
  }
  __syncthreads();

  // ---- Gate math + state update (32 rows x 32 h = 1024 elems, 4 per thread)
  for (int e = tid; e < 1024; e += 256) {
    const int m   = e >> 5;
    const int h   = e & 31;
    const int hg  = hTile * 32 + h;
    const int mg  = mTile * 32 + m;
    const int idx = m * ACCS + h;
    float xi = accLd[0 * ACCG + idx] + bias[0 * HD_ + hg];
    float xf = accLd[1 * ACCG + idx] + bias[1 * HD_ + hg];
    float xo = accLd[2 * ACCG + idx] + bias[2 * HD_ + hg];
    float xc = accLd[3 * ACCG + idx] + bias[3 * HD_ + hg];
    float I  = 1.f / (1.f + __expf(-xi));
    float F  = 1.f / (1.f + __expf(-xf));
    float O  = 1.f / (1.f + __expf(-xo));
    float Ct = tanhf(xc);
    const int sidx = mg * HD_ + hg;
    float Cn = F * Cst[sidx] + I * Ct;
    float Hn = tanhf(Cn) * O;
    out[(size_t)t * B_ * HD_ + sidx] = Hn;
    Cst[sidx]    = Cn;
    Hwrite[sidx] = f32_to_bf16(Hn);
    outH[sidx]   = Hn;   // last step's write survives = final H
    outC[sidx]   = Cn;   // last step's write survives = final C
  }
}

extern "C" void kernel_launch(void* const* d_in, const int* in_sizes, int n_in,
                              void* d_out, int out_size, void* d_ws, size_t ws_size,
                              hipStream_t stream) {
  const float* inputs = (const float*)d_in[0];
  const float* W_xi = (const float*)d_in[1];
  const float* W_hi = (const float*)d_in[2];
  const float* b_i  = (const float*)d_in[3];
  const float* W_xf = (const float*)d_in[4];
  const float* W_hf = (const float*)d_in[5];
  const float* b_f  = (const float*)d_in[6];
  const float* W_xo = (const float*)d_in[7];
  const float* W_ho = (const float*)d_in[8];
  const float* b_o  = (const float*)d_in[9];
  const float* W_xc = (const float*)d_in[10];
  const float* W_hc = (const float*)d_in[11];
  const float* b_c  = (const float*)d_in[12];

  char* ws = (char*)d_ws;
  size_t off = 0;
  auto alloc = [&](size_t bytes) -> void* {
    void* p = ws + off;
    off = (off + bytes + 255) & ~(size_t)255;
    return p;
  };
  unsigned short* Xbf = (unsigned short*)alloc((size_t)T_ * B_ * E_ * 2);  // 32 MB
  unsigned short* Wpk = (unsigned short*)alloc((size_t)KTOT * NTOT * 2);   // 3 MB
  float*          bia = (float*)alloc((size_t)NTOT * 4);
  unsigned short* Hb0 = (unsigned short*)alloc((size_t)B_ * HD_ * 2);
  unsigned short* Hb1 = (unsigned short*)alloc((size_t)B_ * HD_ * 2);
  float*          Cst = (float*)alloc((size_t)B_ * HD_ * 4);

  float* out  = (float*)d_out;
  float* outH = out + (size_t)T_ * B_ * HD_;
  float* outC = outH + (size_t)B_ * HD_;

  pack_weights<<<(KTOT * NTOT + 255) / 256, 256, 0, stream>>>(
      W_xi, W_hi, W_xf, W_hf, W_xo, W_ho, W_xc, W_hc, Wpk);
  pack_bias<<<(NTOT + 255) / 256, 256, 0, stream>>>(b_i, b_f, b_o, b_c, bia);
  cvt_inputs<<<(T_ * B_ * E_ + 255) / 256, 256, 0, stream>>>(inputs, Xbf, T_ * B_ * E_);
  init_state<<<(B_ * HD_ + 255) / 256, 256, 0, stream>>>(Hb0, Hb1, Cst);

  for (int t = 0; t < T_; ++t) {
    const unsigned short* Hr = (t & 1) ? Hb1 : Hb0;
    unsigned short*       Hw = (t & 1) ? Hb0 : Hb1;
    lstm_step<<<dim3(4, 16), 256, 0, stream>>>(
        Xbf, Wpk, bia, Hr, Hw, Cst, out, outH, outC, t);
  }
}
